// NodeTaskHead_78520592105481
// MI455X (gfx1250) — matrix operand order repaired and checked by reading, not compiled, not run on hardware
//
#include <hip/hip_runtime.h>
#include <hip/hip_bf16.h>
#include <stdint.h>

// CDNA5 / gfx1250 — wave32, WMMA bf16 (f32 accumulate)
typedef __attribute__((ext_vector_type(16))) __bf16 v16bf;
typedef __attribute__((ext_vector_type(8)))  __bf16 v8bf;
typedef __attribute__((ext_vector_type(8)))  float  v8f;

#define NNODE 768
#define BSZ   4
#define EMB   512
#define NH    16
#define DH    32
#define MTOT  (BSZ*NNODE)   // 3072 rows

// ---------------------------------------------------------------------------
// k0: x (n_node, bsz, emb) -> out[0..MTOT*EMB) as (bsz, n_node, emb)  (f32 copy)
// ---------------------------------------------------------------------------
__global__ __launch_bounds__(128) void k0_transpose_x(const float* __restrict__ x,
                                                      float* __restrict__ out) {
  int row  = blockIdx.x;               // b*NNODE + node
  int b    = row / NNODE;
  int node = row % NNODE;
  const float4* src = (const float4*)(x + (size_t)(node * BSZ + b) * EMB);
  float4*       dst = (float4*)(out + (size_t)row * EMB);
  dst[threadIdx.x] = src[threadIdx.x];  // 128 threads * 16B = 2KB row
}

// ---------------------------------------------------------------------------
// k1: QKV projection GEMM.  out[m,o] = sum_k xt[m,k] * W[o,k]
//   One wave computes a 16x32 tile = one full head slice (N-pair np == h),
//   reusing the A operand across two WMMAs per K-chunk.
//   Q,K stored (b,H,n,Dh) bf16 (Q pre-scaled); V stored transposed (b,H,Dh,n).
//   Note: NNODE % 16 == 0 -> an M tile never straddles a batch; batch decode
//   and the mat branch are hoisted out of the store loops (no per-element div).
// ---------------------------------------------------------------------------
__global__ __launch_bounds__(128) void k1_qkv_gemm(const float* __restrict__ x,
                                                   const float* __restrict__ Wq,
                                                   const float* __restrict__ Wk,
                                                   const float* __restrict__ Wv,
                                                   __bf16* __restrict__ Qb,
                                                   __bf16* __restrict__ Kb,
                                                   __bf16* __restrict__ Vt) {
  const int wave = threadIdx.x >> 5;
  const int lane = threadIdx.x & 31;
  const int task = blockIdx.x * 4 + wave;   // 0..3071
  const int mt   = task >> 4;               // 0..191 : M tile (16 rows)
  const int np   = task & 15;               // 0..15  : N pair (32 cols) == head
  const int mat  = blockIdx.y;              // 0=Q 1=K 2=V
  const float* W = (mat == 0) ? Wq : (mat == 1) ? Wk : Wv;

  const int ab    = (mt * 16) / NNODE;      // batch (uniform over tile)
  const int node0 = (mt * 16) % NNODE;      // first node of tile

  // A operand: lane holds row m=lane&15; K slots contiguous 16, base 16*(lane>=16)
  const int kbA = (lane >> 4) * 16;
  const float* arow = x + (size_t)((node0 + (lane & 15)) * BSZ + ab) * EMB;

  // B operands: lane holds col n=lane&15; K slots contiguous 16, base 16*(lane>=16)
  const int kbB = (lane >> 4) * 16;
  const float* brow0 = W + (size_t)(np * 32 + (lane & 15)) * EMB;        // dh 0..15
  const float* brow1 = W + (size_t)(np * 32 + 16 + (lane & 15)) * EMB;   // dh 16..31

  v8f acc0 = {}, acc1 = {};
  for (int kc = 0; kc < EMB; kc += 32) {
    v16bf a, b0, b1;
    const float* ap0 = arow + kc + kbA;
    const float* ap1 = arow + kc + kbA + 8;
#pragma unroll
    for (int e = 0; e < 8; ++e) { a[e] = (__bf16)ap0[e]; a[8 + e] = (__bf16)ap1[e]; }
    const float* bp0 = brow0 + kc + kbB;
    const float* bp1 = brow1 + kc + kbB;
#pragma unroll
    for (int e = 0; e < 16; ++e) { b0[e] = (__bf16)bp0[e]; b1[e] = (__bf16)bp1[e]; }
    acc0 = __builtin_amdgcn_wmma_f32_16x16x32_bf16(false, a, false, b0, (short)0,
                                                   acc0, false, false);
    acc1 = __builtin_amdgcn_wmma_f32_16x16x32_bf16(false, a, false, b1, (short)0,
                                                   acc1, false, false);
  }

  // C layout: row = node0 + r + 8*(lane>=16), col(dh) = t*16 + (lane&15)
  const int colr  = lane & 15;
  const int nodeB = node0 + 8 * (lane >> 4);
  const float scaleQ = 0.17677669529663687f;  // 32^-0.5

  if (mat == 0) {
    __bf16* dst = Qb + ((size_t)((ab * NH + np) * NNODE) + nodeB) * DH + colr;
#pragma unroll
    for (int r = 0; r < 8; ++r) {
      dst[(size_t)r * DH]      = (__bf16)(acc0[r] * scaleQ);
      dst[(size_t)r * DH + 16] = (__bf16)(acc1[r] * scaleQ);
    }
  } else if (mat == 1) {
    __bf16* dst = Kb + ((size_t)((ab * NH + np) * NNODE) + nodeB) * DH + colr;
#pragma unroll
    for (int r = 0; r < 8; ++r) {
      dst[(size_t)r * DH]      = (__bf16)acc0[r];
      dst[(size_t)r * DH + 16] = (__bf16)acc1[r];
    }
  } else {
    __bf16* dst = Vt + ((size_t)((ab * NH + np) * DH) + colr) * NNODE + nodeB;
#pragma unroll
    for (int r = 0; r < 8; ++r) {
      dst[r]                       = (__bf16)acc0[r];
      dst[(size_t)16 * NNODE + r]  = (__bf16)acc1[r];
    }
  }
}

// ---------------------------------------------------------------------------
// k2: fused attention. One block = (b, h, 16-row i-tile), 6 waves.
//   phase 1: S(16x768) = q @ K^T via WMMA -> LDS
//   phase 2: softmax rows, folding 1/(dist+1e-4) (and masks) into the rescale
//   phase 3: per wave (c, Dh-half): acc += bf16(P*(pos_c_i - pos_c_j)) @ V
// ---------------------------------------------------------------------------
__global__ __launch_bounds__(192) void k2_attn(const float* __restrict__ pos,
                                               const unsigned char* __restrict__ mask,
                                               const __bf16* __restrict__ Qb,
                                               const __bf16* __restrict__ Kb,
                                               const __bf16* __restrict__ Vt,
                                               float* __restrict__ outv) {
  __shared__ float sS[16 * NNODE];   // 48 KB scores -> probs*inv
  __shared__ float sPx[NNODE], sPy[NNODE], sPz[NNODE], sMsk[NNODE];

  const int it = blockIdx.x;      // 0..47
  const int h  = blockIdx.y;      // 0..15
  const int b  = blockIdx.z;      // 0..3
  const int i0 = it * 16;
  const int tid  = threadIdx.x;
  const int wave = tid >> 5;
  const int lane = tid & 31;

  for (int j = tid; j < NNODE; j += 192) {
    sPx[j]  = pos[(size_t)(b * NNODE + j) * 3 + 0];
    sPy[j]  = pos[(size_t)(b * NNODE + j) * 3 + 1];
    sPz[j]  = pos[(size_t)(b * NNODE + j) * 3 + 2];
    sMsk[j] = mask[b * NNODE + j] ? 1.0f : 0.0f;
  }
  __syncthreads();

  // ---- phase 1: scores ----
  {
    const __bf16* qrow = Qb + ((size_t)((b * NH + h) * NNODE) + i0 + (lane & 15)) * DH;
    const int kbA = (lane >> 4) * 16;
    v8bf qlo = *(const v8bf*)(qrow + kbA);        // 16B aligned
    v8bf qhi = *(const v8bf*)(qrow + kbA + 8);
    v16bf qa = __builtin_shufflevector(qlo, qhi, 0, 1, 2, 3, 4, 5, 6, 7,
                                       8, 9, 10, 11, 12, 13, 14, 15);

    for (int jt = wave; jt < 48; jt += 6) {
      int j0 = jt * 16;
      const __bf16* krow = Kb + ((size_t)((b * NH + h) * NNODE) + j0 + (lane & 15)) * DH
                              + (lane >> 4) * 16;
      v16bf kv = *(const v16bf*)krow;             // 32B aligned
      v8f s = {};
      s = __builtin_amdgcn_wmma_f32_16x16x32_bf16(false, qa, false, kv, (short)0,
                                                  s, false, false);
      int col = j0 + (lane & 15);
      float negadd = (sMsk[col] != 0.0f) ? -3.0e38f : 0.0f;
#pragma unroll
      for (int r = 0; r < 8; ++r) {
        int m = r + 8 * (lane >> 4);
        sS[m * NNODE + col] = s[r] + negadd;
      }
    }
  }
  __syncthreads();

  // ---- phase 2: softmax + fold in delta normalization ----
  if (tid < 128) {
    int row = tid >> 3, sub = tid & 7;   // 8 lanes (same wave) per row
    float* prow = sS + row * NNODE;
    int c0 = sub * 96;
    float mx = -3.4e38f;
    for (int j = 0; j < 96; ++j) mx = fmaxf(mx, prow[c0 + j]);
    mx = fmaxf(mx, __shfl_xor(mx, 1, 32));
    mx = fmaxf(mx, __shfl_xor(mx, 2, 32));
    mx = fmaxf(mx, __shfl_xor(mx, 4, 32));
    float sum = 0.0f;
    for (int j = 0; j < 96; ++j) {
      float e = __expf(prow[c0 + j] - mx);
      prow[c0 + j] = e;
      sum += e;
    }
    sum += __shfl_xor(sum, 1, 32);
    sum += __shfl_xor(sum, 2, 32);
    sum += __shfl_xor(sum, 4, 32);
    float rs = 1.0f / sum;

    float pxi = sPx[i0 + row], pyi = sPy[i0 + row], pzi = sPz[i0 + row];
    float mI  = sMsk[i0 + row];
    for (int j = 0; j < 96; ++j) {
      int col = c0 + j;
      float dx = pxi - sPx[col], dy = pyi - sPy[col], dz = pzi - sPz[col];
      float sq = dx * dx + dy * dy + dz * dz;
      float dist = sq > 0.0f ? sqrtf(sq) : 0.0f;
      if (mI != 0.0f || sMsk[col] != 0.0f) dist = 1e4f;
      float inv = (mI != 0.0f) ? 0.0f : 1.0f / (dist + 1e-4f);
      prow[col] *= rs * inv;     // now holds P * invdist
    }
  }
  __syncthreads();

  // ---- phase 3: vec_c = (P*delta_c) @ V  (wave owns (c, Dh-half)) ----
  {
    const int c  = wave >> 1;    // 0..2
    const int nh = wave & 1;     // 0..1
    const float* sPc = (c == 0) ? sPx : (c == 1) ? sPy : sPz;

    const int mA  = lane & 15;
    const int kbA = (lane >> 4) * 16;
    const float pic = sPc[i0 + mA];
    const float* pS = sS + mA * NNODE;
    const __bf16* vrow = Vt + ((size_t)((b * NH + h) * DH) + nh * 16 + (lane & 15)) * NNODE
                            + (lane >> 4) * 16;
    v8f acc = {};
    for (int kt = 0; kt < 24; ++kt) {   // 768 / 32 K-chunks
      int jb = kt * 32;
      v16bf a;
#pragma unroll
      for (int e = 0; e < 8; ++e) {
        int j0 = jb + kbA + e;
        int j1 = jb + kbA + 8 + e;
        a[e]     = (__bf16)(pS[j0] * (pic - sPc[j0]));
        a[8 + e] = (__bf16)(pS[j1] * (pic - sPc[j1]));
      }
      v16bf bb = *(const v16bf*)(vrow + jb);      // 32B aligned
      acc = __builtin_amdgcn_wmma_f32_16x16x32_bf16(false, a, false, bb, (short)0,
                                                    acc, false, false);
    }
#pragma unroll
    for (int r = 0; r < 8; ++r) {
      int m   = r + 8 * (lane >> 4);
      int col = lane & 15;
      outv[((size_t)((b * NNODE + i0 + m) * 3 + c)) * EMB + h * DH + nh * 16 + col] = acc[r];
    }
  }
}

// ---------------------------------------------------------------------------
extern "C" void kernel_launch(void* const* d_in, const int* in_sizes, int n_in,
                              void* d_out, int out_size, void* d_ws, size_t ws_size,
                              hipStream_t stream) {
  const float* x   = (const float*)d_in[0];
  const float* pos = (const float*)d_in[1];
  const unsigned char* mask = (const unsigned char*)d_in[2];
  const float* Wq  = (const float*)d_in[3];
  const float* Wk  = (const float*)d_in[4];
  const float* Wv  = (const float*)d_in[5];
  float* out = (float*)d_out;

  // Workspace: Q, K (b,H,n,Dh) bf16 + V transposed (b,H,Dh,n) bf16 = 9 MB total
  __bf16* Qb = (__bf16*)d_ws;
  __bf16* Kb = Qb + (size_t)BSZ * NH * NNODE * DH;
  __bf16* Vt = Kb + (size_t)BSZ * NH * NNODE * DH;

  // output #1: x transposed (f32)
  k0_transpose_x<<<MTOT, 128, 0, stream>>>(x, out);

  // QKV projections: 3072 (16x32)-tile tasks / 4 waves per block, y = matrix
  k1_qkv_gemm<<<dim3(768, 3, 1), 128, 0, stream>>>(x, Wq, Wk, Wv, Qb, Kb, Vt);

  // fused attention + position-modulated contraction
  k2_attn<<<dim3(48, NH, BSZ), 192, 0, stream>>>(pos, mask, Qb, Kb, Vt,
                                                 out + (size_t)MTOT * EMB);
}
